// Wafer2SpikeAttention_90108413870090
// MI455X (gfx1250) — matrix operand order repaired and checked by reading, not compiled, not run on hardware
//
#include <hip/hip_runtime.h>
#include <hip/hip_bf16.h>

typedef _Float16 v8h  __attribute__((ext_vector_type(8)));
typedef _Float16 v16h __attribute__((ext_vector_type(16)));
typedef float    v8f  __attribute__((ext_vector_type(8)));

// ---------------------------------------------------------------------------
// WMMA helpers (CDNA5 wave32, v_wmma_f32_16x16x32_f16)
// A fragment: lane L (0-15) holds row M=L, halves {K0..7, K16..23}; lanes 16-31
// hold the same rows with halves {K8..15, K24..31}.  Channel-last activation
// storage makes both 8-half groups contiguous 16B loads.
// B fragment: lane L holds column N=L%16, 16 K halves in A's order, prepacked -> one 32B load.
// One wave computes a 16M x 64N macro-tile: each A fragment is loaded once and
// reused by 4 WMMAs (4 N-tiles), quartering activation traffic.
// ---------------------------------------------------------------------------
__device__ __forceinline__ v16h cat8(v8h lo, v8h hi) {
  v16h r;
#pragma unroll
  for (int i = 0; i < 8; ++i) { r[i] = lo[i]; r[i + 8] = hi[i]; }
  return r;
}

// ---------------------------------------------------------------------------
// Weight packing (dense): f32 [N][K] -> f16 [N/64][K/32][4 nt][32 lanes][16]
// ---------------------------------------------------------------------------
__global__ void pack_dense(const float* __restrict__ src, _Float16* __restrict__ dst,
                           int N, int K, int total) {
  int idx = blockIdx.x * 256 + threadIdx.x;
  if (idx >= total) return;
  int j    = idx & 15;
  int lane = (idx >> 4) & 31;
  int ntl  = (idx >> 9) & 3;
  int rest = idx >> 11;             // ntg*kchunks + kc
  int kchunks = K >> 5;
  int kc  = rest % kchunks;
  int ntg = rest / kchunks;
  int o = (ntg * 4 + ntl) * 16 + (lane & 15);
  int k = kc * 32 + ((lane >> 4) << 3) + (j & 7) + ((j >> 3) << 4);
  dst[idx] = (_Float16)src[(size_t)o * K + k];
}

// conv weights [OC=64][IC=64][7][7] -> [tap49][kc2][nt4][lane32][16]
__global__ void pack_conv(const float* __restrict__ w, _Float16* __restrict__ dst,
                          int total) {
  int idx = blockIdx.x * 256 + threadIdx.x;
  if (idx >= total) return;
  int j    = idx & 15;
  int lane = (idx >> 4) & 31;
  int nt   = (idx >> 9) & 3;
  int kc   = (idx >> 11) & 1;
  int tap  = idx >> 12;             // < 49
  int o = nt * 16 + (lane & 15);
  int c = kc * 32 + ((lane >> 4) << 3) + (j & 7) + ((j >> 3) << 4);
  dst[idx] = (_Float16)w[(size_t)o * 3136 + (size_t)c * 49 + tap];
}

// ---------------------------------------------------------------------------
// conv0: 1->64ch 7x7 stride1 valid, channel-last out [128][30][30][64] (raw)
// ---------------------------------------------------------------------------
__global__ void conv0_kernel(const float* __restrict__ x, const float* __restrict__ w,
                             const float* __restrict__ b, float* __restrict__ z0) {
  int idx = blockIdx.x * 256 + threadIdx.x;   // exactly 7372800 threads
  int c = idx & 63;
  int rest = idx >> 6;
  int xo = rest % 30;
  int rest2 = rest / 30;
  int yo = rest2 % 30;
  int n  = rest2 / 30;
  const float* xin = x + (size_t)n * 36 * 36;
  const float* wc = w + c * 49;
  float acc = b[c];
#pragma unroll
  for (int ky = 0; ky < 7; ++ky)
#pragma unroll
    for (int kx = 0; kx < 7; ++kx)
      acc += wc[ky * 7 + kx] * xin[(yo + ky) * 36 + (xo + kx)];
  z0[idx] = acc;
}

// ---------------------------------------------------------------------------
// GroupNorm (+optional GLIF).  z channel-last [n][P][C]; block = (n,g), 128 thr
// C/P/GROUPS compile-time -> div/mod become shifts / mul-by-reciprocal.
// ---------------------------------------------------------------------------
template <int C, int P, int GROUPS>
__global__ void gn_glif(float* __restrict__ z, const float* __restrict__ gw,
                        const float* __restrict__ gb,
                        float* si, float* sv, float* sa, _Float16* h,
                        int do_glif) {
  constexpr int cpg = C / GROUPS;
  constexpr int elems = cpg * P;
  int g = blockIdx.x % GROUPS;
  int n = blockIdx.x / GROUPS;
  int tid = threadIdx.x;
  __shared__ float rs[128], rq[128];
  float s = 0.f, q = 0.f;
  for (int e = tid; e < elems; e += 128) {
    int cl = e / P, p = e % P;
    size_t idx = ((size_t)n * P + p) * C + (g * cpg + cl);
    float v = z[idx];
    s += v; q += v * v;
  }
  rs[tid] = s; rq[tid] = q;
  __syncthreads();
  for (int st = 64; st > 0; st >>= 1) {
    if (tid < st) { rs[tid] += rs[tid + st]; rq[tid] += rq[tid + st]; }
    __syncthreads();
  }
  constexpr float inv_n = 1.0f / (float)elems;
  float mean = rs[0] * inv_n;
  float var  = rq[0] * inv_n - mean * mean;
  float rstd = rsqrtf(var + 1e-5f);
  for (int e = tid; e < elems; e += 128) {
    int cl = e / P, p = e % P;
    int c = g * cpg + cl;
    size_t idx = ((size_t)n * P + p) * C + c;
    float v = (z[idx] - mean) * rstd * gw[c] + gb[c];
    if (do_glif) {
      float iv = si[idx], vv = sv[idx], av = sa[idx], sp = (float)h[idx];
      iv = 0.5f * iv + v;
      vv = 0.75f * vv * (1.0f - sp) + iv;
      float sn = (vv - (1.0f + 0.1f * av)) > 0.0f ? 1.0f : 0.0f;
      av = 0.9f * av + sn;
      si[idx] = iv; sv[idx] = vv; sa[idx] = av; h[idx] = (_Float16)sn;
    } else {
      z[idx] = v;
    }
  }
}

// GLIF elementwise (layer 0, input z0 precomputed & constant over t)
__global__ void glif_elem(const float* __restrict__ z, float* si, float* sv,
                          float* sa, _Float16* h, int total) {
  int idx = blockIdx.x * 256 + threadIdx.x;
  if (idx >= total) return;
  float iv = si[idx], vv = sv[idx], av = sa[idx];
  float sp = (float)h[idx];
  iv = 0.5f * iv + z[idx];
  vv = 0.75f * vv * (1.0f - sp) + iv;
  float sn = (vv - (1.0f + 0.1f * av)) > 0.0f ? 1.0f : 0.0f;
  av = 0.9f * av + sn;
  si[idx] = iv; sv[idx] = vv; sa[idx] = av; h[idx] = (_Float16)sn;
}

// ---------------------------------------------------------------------------
// Implicit conv GEMM: 7x7 stride2, 64 in-ch, 64 out-ch.
// One wave = 16M x 64N tile (4 fixed accumulator octets; rolled tap loops so
// register allocation stays stable -> no accumulator shuffling movs).
// in channel-last f16 [n][IH][IW][64], z out [Mtotal][64] f32.
// ---------------------------------------------------------------------------
template <int IH, int IW, int OW, int P>
__global__ void conv_gemm(const _Float16* __restrict__ in, const _Float16* __restrict__ wp,
                          const float* __restrict__ bias, float* __restrict__ z) {
  int mt = blockIdx.x;
  int lane = threadIdx.x;
  int ml = lane & 15, kh = (lane >> 4) & 1;
  int m = mt * 16 + ml;
  int n = m / P, p = m % P;
  int oy = p / OW, ox = p % OW;
  const _Float16* base = in + (size_t)n * IH * IW * 64;
  v8f acc[4];
#pragma unroll
  for (int nt = 0; nt < 4; ++nt) {
    float bv = bias[nt * 16 + ml];
#pragma unroll
    for (int r = 0; r < 8; ++r) acc[nt][r] = bv;
  }
  const _Float16* wptr = wp;          // [tap][kc][nt][lane][16]
  int iy0 = 2 * oy, ix0 = 2 * ox;
#pragma unroll 1
  for (int ky = 0; ky < 7; ++ky) {
    const _Float16* rrow = base + (size_t)((iy0 + ky) * IW + ix0) * 64;
#pragma unroll 1
    for (int kx = 0; kx < 7; ++kx) {
      const _Float16* arow = rrow + (size_t)kx * 64;
      __builtin_prefetch(arow + 64, 0, 1);   // -> global_prefetch_b8
#pragma unroll
      for (int kc = 0; kc < 2; ++kc) {
        v8h lo = *(const v8h*)(arow + kc * 32 + kh * 8);
        v8h hi = *(const v8h*)(arow + kc * 32 + kh * 8 + 16);
        v16h Af = cat8(lo, hi);
#pragma unroll
        for (int nt = 0; nt < 4; ++nt) {
          v16h Bf = *(const v16h*)(wptr + (size_t)(kc * 2048 + nt * 512) +
                                   (size_t)lane * 16);
          acc[nt] = __builtin_amdgcn_wmma_f32_16x16x32_f16(false, Af, false, Bf,
                                                           (short)0, acc[nt], false, false);
        }
      }
      wptr += 4096;
    }
  }
#pragma unroll
  for (int nt = 0; nt < 4; ++nt) {
    int col = nt * 16 + ml;
#pragma unroll
    for (int r = 0; r < 8; ++r)
      z[(size_t)(mt * 16 + r + (kh << 3)) * 64 + col] = acc[nt][r];
  }
}

// Dense GEMM: A f16 [M][K] row-major, packed B, f32 out [M][ldo].
// One wave = 16M x 64N (group of 4 N-tiles).
__global__ void dense_gemm4(const _Float16* __restrict__ A, int K,
                            const _Float16* __restrict__ bp,
                            const float* __restrict__ bias,
                            float* __restrict__ out, int ldo, int NTG) {
  int ntg = blockIdx.x % NTG;
  int mt  = blockIdx.x / NTG;
  int lane = threadIdx.x;
  int ml = lane & 15, kh = (lane >> 4) & 1;
  const _Float16* arow = A + (size_t)(mt * 16 + ml) * K;
  int kchunks = K >> 5;
  const _Float16* bpp = bp + (size_t)ntg * kchunks * 2048;
  v8f acc[4];
#pragma unroll
  for (int nt = 0; nt < 4; ++nt) {
    float bv = bias[(ntg * 4 + nt) * 16 + ml];
#pragma unroll
    for (int r = 0; r < 8; ++r) acc[nt][r] = bv;
  }
#pragma unroll 1
  for (int kc = 0; kc < kchunks; ++kc) {
    v8h lo = *(const v8h*)(arow + kc * 32 + kh * 8);
    v8h hi = *(const v8h*)(arow + kc * 32 + kh * 8 + 16);
    v16h Af = cat8(lo, hi);
#pragma unroll
    for (int nt = 0; nt < 4; ++nt) {
      v16h Bf = *(const v16h*)(bpp + (size_t)kc * 2048 + (size_t)nt * 512 +
                               (size_t)lane * 16);
      acc[nt] = __builtin_amdgcn_wmma_f32_16x16x32_f16(false, Af, false, Bf,
                                                       (short)0, acc[nt], false, false);
    }
  }
#pragma unroll
  for (int nt = 0; nt < 4; ++nt) {
    int col = (ntg * 4 + nt) * 16 + ml;
#pragma unroll
    for (int r = 0; r < 8; ++r)
      out[(size_t)(mt * 16 + r + (kh << 3)) * ldo + col] = acc[nt][r];
  }
}

// attn out-proj GEMM (M=1152, K=64, N=64): writes feat f16 with channel-major
// flatten feat[b][e*9+s]
__global__ void proj_gemm(const _Float16* __restrict__ A,
                          const _Float16* __restrict__ bp,
                          const float* __restrict__ bias,
                          _Float16* __restrict__ feat) {
  int mt = blockIdx.x;
  int lane = threadIdx.x;
  int ml = lane & 15, kh = (lane >> 4) & 1;
  const _Float16* arow = A + (size_t)(mt * 16 + ml) * 64;
  v8f acc[4];
#pragma unroll
  for (int nt = 0; nt < 4; ++nt) {
    float bv = bias[nt * 16 + ml];
#pragma unroll
    for (int r = 0; r < 8; ++r) acc[nt][r] = bv;
  }
#pragma unroll
  for (int kc = 0; kc < 2; ++kc) {
    v8h lo = *(const v8h*)(arow + kc * 32 + kh * 8);
    v8h hi = *(const v8h*)(arow + kc * 32 + kh * 8 + 16);
    v16h Af = cat8(lo, hi);
#pragma unroll
    for (int nt = 0; nt < 4; ++nt) {
      v16h Bf = *(const v16h*)(bp + (size_t)(kc * 4 + nt) * 512 + (size_t)lane * 16);
      acc[nt] = __builtin_amdgcn_wmma_f32_16x16x32_f16(false, Af, false, Bf,
                                                       (short)0, acc[nt], false, false);
    }
  }
#pragma unroll
  for (int nt = 0; nt < 4; ++nt) {
    int e = nt * 16 + ml;
#pragma unroll
    for (int r = 0; r < 8; ++r) {
      int m = mt * 16 + r + (kh << 3);
      int b = m / 9, s = m % 9;
      feat[(size_t)b * 576 + e * 9 + s] = (_Float16)acc[nt][r];
    }
  }
}

// ---------------------------------------------------------------------------
// MHA core: per-batch softmax attention (9 tokens, 4 heads, d=16), VALU f32
// ---------------------------------------------------------------------------
__global__ void attn_kernel(const float* __restrict__ qkv, _Float16* __restrict__ ao) {
  int b = blockIdx.x;
  int tid = threadIdx.x;
  if (tid >= 36) return;
  int hh = tid / 9, qi = tid % 9;
  const float* qp = qkv + ((size_t)(b * 9 + qi)) * 192 + hh * 16;
  float sc[9];
  float mx = -1e30f;
  for (int s = 0; s < 9; ++s) {
    const float* kp = qkv + ((size_t)(b * 9 + s)) * 192 + 64 + hh * 16;
    float d = 0.f;
#pragma unroll
    for (int e = 0; e < 16; ++e) d += qp[e] * kp[e];
    d *= 0.25f;               // 1/sqrt(16)
    sc[s] = d;
    mx = fmaxf(mx, d);
  }
  float den = 0.f;
  for (int s = 0; s < 9; ++s) { sc[s] = expf(sc[s] - mx); den += sc[s]; }
  float rden = 1.0f / den;
  for (int e = 0; e < 16; ++e) {
    float o = 0.f;
    for (int s = 0; s < 9; ++s)
      o += sc[s] * qkv[((size_t)(b * 9 + s)) * 192 + 128 + hh * 16 + e];
    ao[((size_t)(b * 9 + qi)) * 64 + hh * 16 + e] = (_Float16)(o * rden);
  }
}

// ---------------------------------------------------------------------------
// head: out[n,k] += (hf . head_w[k] + head_b[k]) * w_t[t]   (spikes are 0/1)
// ---------------------------------------------------------------------------
__global__ void head_kernel(const _Float16* __restrict__ hf, const float* __restrict__ hw,
                            const float* __restrict__ hb, const float* __restrict__ wt,
                            int t, float* __restrict__ out) {
  int n = blockIdx.x, tid = threadIdx.x;
  float acc[9];
#pragma unroll
  for (int k = 0; k < 9; ++k) acc[k] = 0.f;
  for (int j = tid; j < 2304; j += 256) {
    float hv = (float)hf[(size_t)n * 2304 + j];
    if (hv != 0.0f) {
#pragma unroll
      for (int k = 0; k < 9; ++k) acc[k] += hw[k * 2304 + j];
    }
  }
  __shared__ float red[9][256];
#pragma unroll
  for (int k = 0; k < 9; ++k) red[k][tid] = acc[k];
  __syncthreads();
  for (int st = 128; st > 0; st >>= 1) {
    if (tid < st) {
#pragma unroll
      for (int k = 0; k < 9; ++k) red[k][tid] += red[k][tid + st];
    }
    __syncthreads();
  }
  if (tid < 9) out[n * 9 + tid] += (red[tid][0] + hb[tid]) * wt[t];
}

// ---------------------------------------------------------------------------
extern "C" void kernel_launch(void* const* d_in, const int* in_sizes, int n_in,
                              void* d_out, int out_size, void* d_ws, size_t ws_size,
                              hipStream_t stream) {
  const float* x         = (const float*)d_in[0];
  const float* conv0_w   = (const float*)d_in[1];
  const float* conv0_b   = (const float*)d_in[2];
  const float* gn0_w     = (const float*)d_in[3];
  const float* gn0_b     = (const float*)d_in[4];
  const float* conv1_w   = (const float*)d_in[5];
  const float* conv1_b   = (const float*)d_in[6];
  const float* gn1_w     = (const float*)d_in[7];
  const float* gn1_b     = (const float*)d_in[8];
  const float* conv2_w   = (const float*)d_in[9];
  const float* conv2_b   = (const float*)d_in[10];
  const float* gn2_w     = (const float*)d_in[11];
  const float* gn2_b     = (const float*)d_in[12];
  const float* attn_in_w = (const float*)d_in[13];
  const float* attn_in_b = (const float*)d_in[14];
  const float* attn_out_w= (const float*)d_in[15];
  const float* attn_out_b= (const float*)d_in[16];
  const float* fc_w      = (const float*)d_in[17];
  const float* fc_b      = (const float*)d_in[18];
  const float* gnfc_w    = (const float*)d_in[19];
  const float* gnfc_b    = (const float*)d_in[20];
  const float* head_w    = (const float*)d_in[21];
  const float* head_b    = (const float*)d_in[22];
  const float* w_t       = (const float*)d_in[23];
  float* out = (float*)d_out;

  const int E0 = 128 * 30 * 30 * 64;   // 7,372,800
  const int E1 = 128 * 12 * 12 * 64;   // 1,179,648
  const int E2 = 128 * 9 * 64;         // 73,728
  const int EQ = 128 * 9 * 192;        // 221,184
  const int EF = 128 * 2304;           // 294,912

  char* base = (char*)d_ws;
  size_t off = 0;
  auto alloc = [&](size_t bytes) -> char* {
    off = (off + 255) & ~(size_t)255;
    char* p = base + off;
    off += bytes;
    return p;
  };

  float*    z0  = (float*)alloc((size_t)E0 * 4);
  float*    i0  = (float*)alloc((size_t)E0 * 4);
  float*    v0  = (float*)alloc((size_t)E0 * 4);
  float*    a0  = (float*)alloc((size_t)E0 * 4);
  _Float16* h0  = (_Float16*)alloc((size_t)E0 * 2);
  float*    z1  = (float*)alloc((size_t)E1 * 4);
  float*    i1  = (float*)alloc((size_t)E1 * 4);
  float*    v1  = (float*)alloc((size_t)E1 * 4);
  float*    a1  = (float*)alloc((size_t)E1 * 4);
  _Float16* h1  = (_Float16*)alloc((size_t)E1 * 2);
  float*    z2  = (float*)alloc((size_t)E2 * 4);
  float*    i2  = (float*)alloc((size_t)E2 * 4);
  float*    v2  = (float*)alloc((size_t)E2 * 4);
  float*    a2  = (float*)alloc((size_t)E2 * 4);
  _Float16* h2  = (_Float16*)alloc((size_t)E2 * 2);
  float*    qkv = (float*)alloc((size_t)EQ * 4);
  _Float16* ao  = (_Float16*)alloc((size_t)E2 * 2);
  _Float16* ft  = (_Float16*)alloc((size_t)E2 * 2);
  float*    zfc = (float*)alloc((size_t)EF * 4);
  float*    ifc = (float*)alloc((size_t)EF * 4);
  float*    vfc = (float*)alloc((size_t)EF * 4);
  float*    afc = (float*)alloc((size_t)EF * 4);
  _Float16* hfc = (_Float16*)alloc((size_t)EF * 2);
  _Float16* w1p = (_Float16*)alloc((size_t)200704 * 2);
  _Float16* w2p = (_Float16*)alloc((size_t)200704 * 2);
  _Float16* qip = (_Float16*)alloc((size_t)12288 * 2);
  _Float16* qop = (_Float16*)alloc((size_t)4096 * 2);
  _Float16* fcp = (_Float16*)alloc((size_t)1327104 * 2);
  (void)ws_size; (void)in_sizes; (void)n_in; (void)out_size;

  // zero GLIF state + spike buffers + output accumulator (deterministic per call)
  hipMemsetAsync(i0, 0, (size_t)E0 * 12, stream);     // i0,v0,a0 contiguous
  hipMemsetAsync(h0, 0, (size_t)E0 * 2, stream);
  hipMemsetAsync(i1, 0, (size_t)E1 * 12, stream);
  hipMemsetAsync(h1, 0, (size_t)E1 * 2, stream);
  hipMemsetAsync(i2, 0, (size_t)E2 * 12, stream);
  hipMemsetAsync(h2, 0, (size_t)E2 * 2, stream);
  hipMemsetAsync(ifc, 0, (size_t)EF * 12, stream);
  hipMemsetAsync(hfc, 0, (size_t)EF * 2, stream);
  hipMemsetAsync(out, 0, (size_t)128 * 9 * 4, stream);

  // pack weights to f16 WMMA-B layout
  pack_conv<<<784, 256, 0, stream>>>(conv1_w, w1p, 200704);
  pack_conv<<<784, 256, 0, stream>>>(conv2_w, w2p, 200704);
  pack_dense<<<48, 256, 0, stream>>>(attn_in_w, qip, 192, 64, 12288);
  pack_dense<<<16, 256, 0, stream>>>(attn_out_w, qop, 64, 64, 4096);
  pack_dense<<<5184, 256, 0, stream>>>(fc_w, fcp, 2304, 576, 1327104);

  // conv0 + GN0 (constant across timesteps): z0 normalized in place
  conv0_kernel<<<28800, 256, 0, stream>>>(x, conv0_w, conv0_b, z0);
  gn_glif<64, 900, 8><<<128 * 8, 128, 0, stream>>>(z0, gn0_w, gn0_b,
                                                   nullptr, nullptr, nullptr, nullptr, 0);

  for (int t = 0; t < 8; ++t) {
    glif_elem<<<28800, 256, 0, stream>>>(z0, i0, v0, a0, h0, E0);
    conv_gemm<30, 30, 12, 144><<<1152, 32, 0, stream>>>(h0, w1p, conv1_b, z1);
    gn_glif<64, 144, 8><<<128 * 8, 128, 0, stream>>>(z1, gn1_w, gn1_b,
                                                     i1, v1, a1, h1, 1);
    conv_gemm<12, 12, 3, 9><<<72, 32, 0, stream>>>(h1, w2p, conv2_b, z2);
    gn_glif<64, 9, 8><<<128 * 8, 128, 0, stream>>>(z2, gn2_w, gn2_b,
                                                   i2, v2, a2, h2, 1);
    dense_gemm4<<<72 * 3, 32, 0, stream>>>(h2, 64, qip, attn_in_b, qkv, 192, 3);
    attn_kernel<<<128, 64, 0, stream>>>(qkv, ao);
    proj_gemm<<<72, 32, 0, stream>>>(ao, qop, attn_out_b, ft);
    dense_gemm4<<<8 * 36, 32, 0, stream>>>(ft, 576, fcp, fc_b, zfc, 2304, 36);
    gn_glif<2304, 1, 32><<<128 * 32, 128, 0, stream>>>(zfc, gnfc_w, gnfc_b,
                                                       ifc, vfc, afc, hfc, 1);
    head_kernel<<<128, 256, 0, stream>>>(hfc, head_w, head_b, w_t, t, out);
  }
}
